// Combiner_73409581023633
// MI455X (gfx1250) — matrix operand order repaired and checked by reading, not compiled, not run on hardware
//
#include <hip/hip_runtime.h>

// ---------------------------------------------------------------------------
// Sizes from the reference
// ---------------------------------------------------------------------------
#define B_    256
#define SEQ_  256
#define LAT_  512
#define HID_  1024
#define BT    16            // batch rows per workgroup (WMMA M)
#define ZPAD  (LAT_ + 8)    // padded LDS row strides (bf16 elems) to spread banks
#define HPAD  (HID_ + 8)

typedef __attribute__((ext_vector_type(16))) __bf16 v16bf;
typedef __attribute__((ext_vector_type(8)))  __bf16 v8bf;
typedef __attribute__((ext_vector_type(8)))  float  v8f;

// float -> bf16 bits, round-to-nearest-even (bit-level; avoids host __bf16 use)
__device__ __forceinline__ unsigned short f2bf_bits(float f) {
    unsigned u = __builtin_bit_cast(unsigned, f);
    unsigned r = (u + 0x7FFFu + ((u >> 16) & 1u)) >> 16;
    return (unsigned short)r;
}

// Branch-free fast tanh: 1 - 2/(e^{2x}+1).
// e^{2x} -> inf gives 1.0; e^{2x} -> 0 gives -1.0; no divergence.
__device__ __forceinline__ float tanh_fast(float x) {
    float e = __expf(2.0f * x);                 // v_exp_f32
    return 1.0f - 2.0f * __builtin_amdgcn_rcpf(e + 1.0f);
}

// Branch-free stable softplus: max(x,0) + log(1 + e^{-|x|}).
__device__ __forceinline__ float softplus_fast(float x) {
    return fmaxf(x, 0.0f) + __logf(1.0f + __expf(-fabsf(x)));
}

// ---------------------------------------------------------------------------
// Prologue: pack a K x N f32 row-major weight matrix into bf16 WMMA B-fragments.
// Fragment (nt, kb) covers cols [nt*16, nt*16+16), K [kb*32, kb*32+32).
// B layout used here (contiguous k, same as the A reads): lane L holds col N = nt*16 + (L&15), K-range
// kb*32 + (L>>4)*16 .. +15, 16 contiguous bf16 -> 32 bytes per lane.
// Fragment index = nt*(K/32) + kb, 512 elems each.
// ---------------------------------------------------------------------------
__global__ void pack_wb(const float* __restrict__ src,
                        unsigned short* __restrict__ dst,
                        int K, int N) {
    const int wave = threadIdx.x >> 5;
    const int lane = threadIdx.x & 31;
    const int KB   = K >> 5;
    const int frag = blockIdx.x * (blockDim.x >> 5) + wave;
    if (frag >= (N >> 4) * KB) return;
    const int nt = frag / KB;
    const int kb = frag - nt * KB;
    const int n  = nt * 16 + (lane & 15);
    const int k0 = kb * 32 + (lane >> 4) * 16;
    unsigned short* out = dst + (size_t)frag * 512 + lane * 16;
    #pragma unroll
    for (int i = 0; i < 16; ++i)
        out[i] = f2bf_bits(src[(size_t)(k0 + i) * N + n]);
}

// ---------------------------------------------------------------------------
// Persistent recurrence kernel: one workgroup per 16 batch rows, 16 waves.
// ---------------------------------------------------------------------------
__global__ void __launch_bounds__(512)
vrnn_kernel(const float* __restrict__ h_right,   // [B, SEQ, HID]
            const float* __restrict__ z_init,    // [B, LAT]
            const float* __restrict__ eps,       // [SEQ, B, LAT]
            const float* __restrict__ b_c,       // [HID]
            const float* __restrict__ b_mu,      // [LAT]
            const float* __restrict__ b_sig,     // [LAT]
            const unsigned short* __restrict__ wc_p,   // packed bf16 frags
            const unsigned short* __restrict__ wmu_p,
            const unsigned short* __restrict__ wsig_p,
            float* __restrict__ out_z,
            float* __restrict__ out_mu,
            float* __restrict__ out_sig)
{
    __shared__ unsigned short zs[BT * ZPAD];   // z_{t-1} tile, bf16 bits
    __shared__ unsigned short hs[BT * HPAD];   // h tile, bf16 bits

    const int tid  = threadIdx.x;
    const int wave = tid >> 5;
    const int lane = tid & 31;
    const int half = lane >> 4;        // 0: lanes 0-15, 1: lanes 16-31
    const int lc   = lane & 15;
    const int b0   = blockIdx.x * BT;

    // Loop-invariant biases for this wave's columns (hoisted out of t-loop)
    float bcv[4];
    #pragma unroll
    for (int i = 0; i < 4; ++i) bcv[i] = b_c[wave * 64 + i * 16 + lc];
    float bmuv[2], bsgv[2];
    #pragma unroll
    for (int i = 0; i < 2; ++i) {
        bmuv[i] = b_mu[wave * 32 + i * 16 + lc];
        bsgv[i] = b_sig[wave * 32 + i * 16 + lc];
    }

    // Load initial z into LDS (bf16)
    for (int idx = tid; idx < BT * LAT_; idx += 512) {
        const int r = idx >> 9;        // / 512
        const int c = idx & (LAT_ - 1);
        zs[r * ZPAD + c] = f2bf_bits(z_init[(size_t)(b0 + r) * LAT_ + c]);
    }
    __syncthreads();

    for (int t = 0; t < SEQ_; ++t) {
        // ---------------- Phase A: h = 0.5*(tanh(z @ W_c + b_c) + h_r) -----
        // 16 waves x 64 HID columns each (4 WMMA N-tiles)
        const int colA = wave * 64;
        #pragma unroll
        for (int nt4 = 0; nt4 < 4; ++nt4) {
            const int col0 = colA + nt4 * 16;
            const unsigned short* brow =
                wc_p + ((size_t)(col0 >> 4) * (LAT_ >> 5)) * 512 + lane * 16;
            v8f acc = {};
            #pragma unroll 4
            for (int kb = 0; kb < (LAT_ >> 5); ++kb) {
                // A fragment from z LDS: two 16B chunks per lane
                const unsigned short* zp = &zs[lc * ZPAD + kb * 32 + half * 16];
                v8bf alo = *(const v8bf*)zp;
                v8bf ahi = *(const v8bf*)(zp + 8);
                v16bf a = __builtin_shufflevector(
                    alo, ahi, 0,1,2,3,4,5,6,7,8,9,10,11,12,13,14,15);
                v16bf b = *(const v16bf*)(brow + (size_t)kb * 512);
                acc = __builtin_amdgcn_wmma_f32_16x16x32_bf16(
                          false, a, false, b, (short)0, acc, false, false);
            }
            const int   n  = col0 + lc;
            const float bc = bcv[nt4];
            #pragma unroll
            for (int r = 0; r < 8; ++r) {
                const int   m  = r + half * 8;
                const float hr = h_right[((size_t)(b0 + m) * SEQ_ + t) * HID_ + n];
                const float hv = 0.5f * (tanh_fast(acc[r] + bc) + hr);
                hs[m * HPAD + n] = f2bf_bits(hv);
            }
        }
        __syncthreads();

        // ---------------- Phase B: mu / sigma / z --------------------------
        // 16 waves x 32 LAT columns each; both GEMMs share the A fragments.
        const int colB = wave * 32;
        #pragma unroll
        for (int nt2 = 0; nt2 < 2; ++nt2) {
            const int col0 = colB + nt2 * 16;
            const size_t fb = (size_t)(col0 >> 4) * (HID_ >> 5) * 512;
            const unsigned short* bmrow = wmu_p  + fb + lane * 16;
            const unsigned short* bsrow = wsig_p + fb + lane * 16;
            v8f amu = {}, asg = {};
            #pragma unroll 4
            for (int kb = 0; kb < (HID_ >> 5); ++kb) {
                const unsigned short* hp = &hs[lc * HPAD + kb * 32 + half * 16];
                v8bf alo = *(const v8bf*)hp;
                v8bf ahi = *(const v8bf*)(hp + 8);
                v16bf a = __builtin_shufflevector(
                    alo, ahi, 0,1,2,3,4,5,6,7,8,9,10,11,12,13,14,15);
                v16bf bm = *(const v16bf*)(bmrow + (size_t)kb * 512);
                v16bf bs = *(const v16bf*)(bsrow + (size_t)kb * 512);
                amu = __builtin_amdgcn_wmma_f32_16x16x32_bf16(
                          false, a, false, bm, (short)0, amu, false, false);
                asg = __builtin_amdgcn_wmma_f32_16x16x32_bf16(
                          false, a, false, bs, (short)0, asg, false, false);
            }
            const int   n   = col0 + lc;
            const float bmu = bmuv[nt2];
            const float bsg = bsgv[nt2];
            #pragma unroll
            for (int r = 0; r < 8; ++r) {
                const int    m  = r + half * 8;
                const float  mu = amu[r] + bmu;
                const float  sg = softplus_fast(asg[r] + bsg);
                const float  e  = eps[((size_t)t * B_ + (b0 + m)) * LAT_ + n];
                const float  z  = mu + sg * e;
                const size_t o  = ((size_t)(b0 + m) * SEQ_ + t) * LAT_ + n;
                out_z[o]   = z;
                out_mu[o]  = mu;
                out_sig[o] = sg;
                zs[m * ZPAD + n] = f2bf_bits(z);   // feed next timestep
            }
        }
        __syncthreads();   // protect zs (next Phase A) and hs (next rewrite)
    }
}

// ---------------------------------------------------------------------------
// Launch
// ---------------------------------------------------------------------------
extern "C" void kernel_launch(void* const* d_in, const int* in_sizes, int n_in,
                              void* d_out, int out_size, void* d_ws, size_t ws_size,
                              hipStream_t stream) {
    const float* h_right = (const float*)d_in[0];
    const float* z_init  = (const float*)d_in[1];
    const float* eps     = (const float*)d_in[2];
    const float* W_c     = (const float*)d_in[3];
    const float* b_c     = (const float*)d_in[4];
    const float* W_mu    = (const float*)d_in[5];
    const float* b_mu    = (const float*)d_in[6];
    const float* W_sig   = (const float*)d_in[7];
    const float* b_sig   = (const float*)d_in[8];

    // Workspace: packed bf16 weights, 3 MB total
    unsigned short* wc_p   = (unsigned short*)d_ws;
    unsigned short* wmu_p  = wc_p  + (size_t)LAT_ * HID_;
    unsigned short* wsig_p = wmu_p + (size_t)HID_ * LAT_;

    float* out_z   = (float*)d_out;
    float* out_mu  = out_z  + (size_t)B_ * SEQ_ * LAT_;
    float* out_sig = out_mu + (size_t)B_ * SEQ_ * LAT_;

    // Pack weights into WMMA B-fragment order (8 fragments per 256-thread block)
    const int fragsC = (HID_ / 16) * (LAT_ / 32);   // 1024
    const int fragsM = (LAT_ / 16) * (HID_ / 32);   // 1024
    pack_wb<<<(fragsC + 7) / 8, 256, 0, stream>>>(W_c,   wc_p,   LAT_, HID_);
    pack_wb<<<(fragsM + 7) / 8, 256, 0, stream>>>(W_mu,  wmu_p,  HID_, LAT_);
    pack_wb<<<(fragsM + 7) / 8, 256, 0, stream>>>(W_sig, wsig_p, HID_, LAT_);

    // Persistent recurrence: one WG per 16 batch rows, 16 waves each
    vrnn_kernel<<<B_ / BT, 512, 0, stream>>>(
        h_right, z_init, eps, b_c, b_mu, b_sig,
        wc_p, wmu_p, wsig_p, out_z, out_mu, out_sig);
}
